// Q_RN07_62165356642832
// MI455X (gfx1250) — 8-bit matrix operand order repaired and checked by reading, compiled, not run on hardware
//
#include <hip/hip_runtime.h>
#include <stdint.h>

typedef int v8i __attribute__((ext_vector_type(8)));

#define OFF_SLOTS   0u
#define OFF_AB      256u
#define OFF_BPK     512u
#define OFF_Z       66048u
#define OFF_ACTQ    131584u
#define OFF_ACTF    (OFF_ACTQ + 33554432u)
#define OFF_APK     (OFF_ACTF + 134217728u)

__device__ __forceinline__ float sf_from_bits(unsigned bits) {
  return fmaxf(__uint_as_float(bits) * (1.0f / 127.0f), 1e-8f);
}

__device__ __forceinline__ int q8(float v, float inv) {
  float r = rintf(v * inv);
  r = fminf(fmaxf(r, -127.0f), 127.0f);
  return (int)r;
}

__global__ void k_init(unsigned* slots, int n) {
  int i = blockIdx.x * blockDim.x + threadIdx.x;
  if (i < n) slots[i] = 0u;
}

__global__ void __launch_bounds__(256) k_absmax4(const float4* __restrict__ x,
                                                 long long n4, unsigned* slot) {
  __shared__ float sm[256];
  float m = 0.0f;
  long long stride = (long long)gridDim.x * blockDim.x;
  for (long long i = (long long)blockIdx.x * blockDim.x + threadIdx.x; i < n4; i += stride) {
    float4 v = x[i];
    m = fmaxf(m, fmaxf(fmaxf(fabsf(v.x), fabsf(v.y)), fmaxf(fabsf(v.z), fabsf(v.w))));
  }
  sm[threadIdx.x] = m;
  __syncthreads();
  for (int s = 128; s > 0; s >>= 1) {
    if ((int)threadIdx.x < s) sm[threadIdx.x] = fmaxf(sm[threadIdx.x], sm[threadIdx.x + s]);
    __syncthreads();
  }
  if (threadIdx.x == 0) atomicMax(slot, __float_as_uint(sm[0]));
}

__global__ void __launch_bounds__(256) k_quant4(const float4* __restrict__ x,
                                                unsigned* __restrict__ q,
                                                long long n4,
                                                const unsigned* __restrict__ slot) {
  float inv = 1.0f / sf_from_bits(*slot);
  long long stride = (long long)gridDim.x * blockDim.x;
  for (long long i = (long long)blockIdx.x * blockDim.x + threadIdx.x; i < n4; i += stride) {
    float4 v = x[i];
    unsigned p = (unsigned)(unsigned char)(int8_t)q8(v.x, inv)
               | ((unsigned)(unsigned char)(int8_t)q8(v.y, inv) << 8)
               | ((unsigned)(unsigned char)(int8_t)q8(v.z, inv) << 16)
               | ((unsigned)(unsigned char)(int8_t)q8(v.w, inv) << 24);
    q[i] = p;
  }
}

__global__ void k_prep(const float* __restrict__ w, const float* __restrict__ bias,
                       const float* g, const float* bb, const float* mu, const float* var,
                       int has_bn, int OC, int K, int Kblocks, int NT,
                       const unsigned* __restrict__ slot,
                       unsigned* __restrict__ Bpk, float* __restrict__ ab) {
  __shared__ float s_invwsf[32];
  float sf_prev = sf_from_bits(*slot);
  int t = threadIdx.x;
  if (t < OC) {
    const float* row = w + (long long)t * K;
    float m = 0.0f;
    for (int k = 0; k < K; ++k) m = fmaxf(m, fabsf(row[k]));
    float wsf = fmaxf(m * (1.0f / 127.0f), 1e-8f);
    s_invwsf[t] = 1.0f / wsf;
    float bsf  = wsf * sf_prev;
    float bint = rintf(bias[t] / bsf);
    float alpha, beta;
    if (has_bn) {
      float inv   = g[t] * rsqrtf(var[t] + 1e-5f);
      float shift = bb[t] - mu[t] * inv;
      alpha = bsf * inv;
      beta  = bint * bsf * inv + shift;
    } else {
      alpha = bsf;
      beta  = bint * bsf;
    }
    ab[t]      = alpha;
    ab[32 + t] = beta;
  }
  __syncthreads();
  int total = Kblocks * NT * 256;
  for (int d = t; d < total; d += blockDim.x) {
    int tile = d >> 8;
    int lane = (d >> 3) & 31;
    int j    = d & 7;
    int kb   = tile / NT, nt = tile % NT;
    int n    = nt * 16 + (lane & 15);
    int half = lane >> 4;
    int kbase = kb * 64 + ((j >> 2) * 32) + half * 16 + (j & 3) * 4;
    unsigned v = 0u;
    if (n < OC) {
      const float* row = w + (long long)n * K;
      float iw = s_invwsf[n];
      for (int bb4 = 0; bb4 < 4; ++bb4) {
        int k = kbase + bb4;
        if (k < K) {
          int qv = (int)rintf(row[k] * iw);
          qv = qv < -127 ? -127 : (qv > 127 ? 127 : qv);
          v |= (unsigned)(unsigned char)(int8_t)qv << (8 * bb4);
        }
      }
    }
    Bpk[d] = v;
  }
}

template <int C, int H, int W, int KH, int KW, int ST, int PAD, int OH, int OW, int KB>
__global__ void __launch_bounds__(256) k_pack_t(const int8_t* __restrict__ aq,
                                                unsigned* __restrict__ Apk,
                                                int Mtiles) {
  int d = blockIdx.x * blockDim.x + threadIdx.x;
  int total = Mtiles * KB * 256;
  if (d >= total) return;
  int tile = d >> 8;
  int lane = (d >> 3) & 31;
  int j    = d & 7;
  int mt = tile / KB, kb = tile % KB;
  int m    = mt * 16 + (lane & 15);
  int half = lane >> 4;
  int kbase = kb * 64 + ((j >> 2) * 32) + half * 16 + (j & 3) * 4;
  constexpr int SPATIAL = OH * OW;
  constexpr int KK      = C * KH * KW;
  int img = m / SPATIAL;
  int rem = m - img * SPATIAL;
  int oy  = rem / OW;
  int ox  = rem - oy * OW;
  unsigned v = 0u;
#pragma unroll
  for (int bb4 = 0; bb4 < 4; ++bb4) {
    int k = kbase + bb4;
    if (k < KK) {
      int c  = k / (KH * KW);
      int r2 = k - c * (KH * KW);
      int ky = r2 / KW, kx = r2 - ky * KW;
      int iy = oy * ST - PAD + ky;
      int ix = ox * ST - PAD + kx;
      if (iy >= 0 && iy < H && ix >= 0 && ix < W) {
        int8_t b = aq[(((long long)img * C + c) * H + iy) * W + ix];
        v |= (unsigned)(unsigned char)b << (8 * bb4);
      }
    }
  }
  Apk[d] = v;
}

template <int KB, int NT, int OC, int SPATIAL>
__global__ void __launch_bounds__(256) k_gemm_t(const unsigned* __restrict__ Apk,
                                                const unsigned* __restrict__ Bpk,
                                                const float* __restrict__ ab,
                                                float* __restrict__ out,
                                                int Mtiles) {
  int wave = threadIdx.x >> 5;
  int lane = threadIdx.x & 31;
  int mt = blockIdx.x * 8 + wave;
  if (mt >= Mtiles) return;

  const v8i* A = (const v8i*)Apk + (size_t)mt * KB * 32 + lane;
  const v8i* B = (const v8i*)Bpk + lane;
  v8i acc0 = {};
  v8i acc1 = {};
#pragma unroll(KB <= 8 ? KB : 4)
  for (int kb = 0; kb < KB; ++kb) {
    v8i a  = A[kb * 32];
    v8i b0 = B[(kb * NT + 0) * 32];
    acc0 = __builtin_amdgcn_wmma_i32_16x16x64_iu8(true, a, true, b0, acc0, false, false);
    if constexpr (NT > 1) {
      v8i b1 = B[(kb * NT + 1) * 32];
      acc1 = __builtin_amdgcn_wmma_i32_16x16x64_iu8(true, a, true, b1, acc1, false, false);
    }
  }

  int half = lane >> 4;
  int ncol = lane & 15;
#pragma unroll
  for (int nt = 0; nt < NT; ++nt) {
    v8i acc;
    if (nt == 0) acc = acc0; else acc = acc1;
    int n = nt * 16 + ncol;
    if (n >= OC) continue;
    float alpha = ab[n], beta = ab[32 + n];
    if constexpr (SPATIAL % 16 == 0) {
      int img = (mt * 16) / SPATIAL;
      int rem = mt * 16 - img * SPATIAL + half * 8;
      float* dst = out + ((long long)img * OC + n) * SPATIAL + rem;
      float4 lo, hi;
      lo.x = (float)acc[0] * alpha + beta;
      lo.y = (float)acc[1] * alpha + beta;
      lo.z = (float)acc[2] * alpha + beta;
      lo.w = (float)acc[3] * alpha + beta;
      hi.x = (float)acc[4] * alpha + beta;
      hi.y = (float)acc[5] * alpha + beta;
      hi.z = (float)acc[6] * alpha + beta;
      hi.w = (float)acc[7] * alpha + beta;
      ((float4*)dst)[0] = lo;
      ((float4*)dst)[1] = hi;
    } else {
      int mrow_base = mt * 16 + half * 8;
#pragma unroll
      for (int r = 0; r < 8; ++r) {
        int m = mrow_base + r;
        out[(long long)m * OC + n] = (float)acc[r] * alpha + beta;
      }
    }
  }
}

__global__ void __launch_bounds__(256) k_softmax(const float* __restrict__ z,
                                                 float* __restrict__ out,
                                                 int B, int NC) {
  int col = blockIdx.x;
  __shared__ float sm[256];
  float m = -3.4e38f;
  for (int r = threadIdx.x; r < B; r += blockDim.x) m = fmaxf(m, z[(long long)r * NC + col]);
  sm[threadIdx.x] = m;
  __syncthreads();
  for (int s = 128; s > 0; s >>= 1) {
    if ((int)threadIdx.x < s) sm[threadIdx.x] = fmaxf(sm[threadIdx.x], sm[threadIdx.x + s]);
    __syncthreads();
  }
  m = sm[0];
  __syncthreads();
  float sum = 0.0f;
  for (int r = threadIdx.x; r < B; r += blockDim.x) sum += expf(z[(long long)r * NC + col] - m);
  sm[threadIdx.x] = sum;
  __syncthreads();
  for (int s = 128; s > 0; s >>= 1) {
    if ((int)threadIdx.x < s) sm[threadIdx.x] += sm[threadIdx.x + s];
    __syncthreads();
  }
  float inv = 1.0f / sm[0];
  for (int r = threadIdx.x; r < B; r += blockDim.x)
    out[(long long)r * NC + col] = expf(z[(long long)r * NC + col] - m) * inv;
}

static inline int red_grid(long long n4) {
  long long g = (n4 + 255) / 256;
  return (int)(g < 4096 ? g : 4096);
}

extern "C" void kernel_launch(void* const* d_in, const int* in_sizes, int n_in,
                              void* d_out, int out_size, void* d_ws, size_t ws_size,
                              hipStream_t stream) {
  (void)in_sizes; (void)n_in; (void)out_size; (void)ws_size;

  const float* x = (const float*)d_in[0];
  const float *W[5], *Bi[5], *G[5], *BB[5], *MU[5], *VAR[5];
  for (int i = 0; i < 5; ++i) {
    int b0 = 1 + 6 * i;
    W[i]   = (const float*)d_in[b0 + 0];
    Bi[i]  = (const float*)d_in[b0 + 1];
    G[i]   = (const float*)d_in[b0 + 2];
    BB[i]  = (const float*)d_in[b0 + 3];
    MU[i]  = (const float*)d_in[b0 + 4];
    VAR[i] = (const float*)d_in[b0 + 5];
  }
  const float* wd = (const float*)d_in[31];
  const float* bd = (const float*)d_in[32];

  uint8_t* ws = (uint8_t*)d_ws;
  unsigned* slots = (unsigned*)(ws + OFF_SLOTS);
  float*    ab    = (float*)(ws + OFF_AB);
  unsigned* Bpk   = (unsigned*)(ws + OFF_BPK);
  float*    zbuf  = (float*)(ws + OFF_Z);
  int8_t*   actq  = (int8_t*)(ws + OFF_ACTQ);
  float*    actf  = (float*)(ws + OFF_ACTF);
  unsigned* Apk   = (unsigned*)(ws + OFF_APK);

  k_init<<<1, 32, 0, stream>>>(slots, 8);

  long long nIn4 = 1024LL * 3 * 32 * 32 / 4;
  k_absmax4<<<red_grid(nIn4), 256, 0, stream>>>((const float4*)x, nIn4, slots + 0);
  k_quant4 <<<red_grid(nIn4), 256, 0, stream>>>((const float4*)x, (unsigned*)actq, nIn4, slots + 0);

  k_prep<<<1, 256, 0, stream>>>(W[0], Bi[0], G[0], BB[0], MU[0], VAR[0],
                                1, 32, 27, 1, 2, slots + 0, Bpk, ab);
  k_pack_t<3, 32, 32, 3, 3, 1, 1, 32, 32, 1><<<65536, 256, 0, stream>>>(actq, Apk, 65536);
  k_gemm_t<1, 2, 32, 1024><<<8192, 256, 0, stream>>>(Apk, Bpk, ab, actf, 65536);
  long long n0_4 = 1024LL * 32 * 32 * 32 / 4;
  k_absmax4<<<red_grid(n0_4), 256, 0, stream>>>((const float4*)actf, n0_4, slots + 1);
  k_quant4 <<<red_grid(n0_4), 256, 0, stream>>>((const float4*)actf, (unsigned*)actq, n0_4, slots + 1);

  k_prep<<<1, 256, 0, stream>>>(W[1], Bi[1], G[1], BB[1], MU[1], VAR[1],
                                1, 4, 32, 1, 1, slots + 1, Bpk, ab);
  k_pack_t<32, 32, 32, 1, 1, 1, 0, 32, 32, 1><<<65536, 256, 0, stream>>>(actq, Apk, 65536);
  k_gemm_t<1, 1, 4, 1024><<<8192, 256, 0, stream>>>(Apk, Bpk, ab, actf, 65536);
  long long n1_4 = 1024LL * 4 * 32 * 32 / 4;
  k_absmax4<<<red_grid(n1_4), 256, 0, stream>>>((const float4*)actf, n1_4, slots + 2);
  k_quant4 <<<red_grid(n1_4), 256, 0, stream>>>((const float4*)actf, (unsigned*)actq, n1_4, slots + 2);

  k_prep<<<1, 256, 0, stream>>>(W[2], Bi[2], G[2], BB[2], MU[2], VAR[2],
                                1, 32, 36, 1, 2, slots + 2, Bpk, ab);
  k_pack_t<4, 32, 32, 3, 3, 2, 1, 16, 16, 1><<<16384, 256, 0, stream>>>(actq, Apk, 16384);
  k_gemm_t<1, 2, 32, 256><<<2048, 256, 0, stream>>>(Apk, Bpk, ab, actf, 16384);
  long long n2_4 = 1024LL * 32 * 16 * 16 / 4;
  k_absmax4<<<red_grid(n2_4), 256, 0, stream>>>((const float4*)actf, n2_4, slots + 3);
  k_quant4 <<<red_grid(n2_4), 256, 0, stream>>>((const float4*)actf, (unsigned*)actq, n2_4, slots + 3);

  k_prep<<<1, 256, 0, stream>>>(W[3], Bi[3], G[3], BB[3], MU[3], VAR[3],
                                1, 32, 288, 5, 2, slots + 3, Bpk, ab);
  k_pack_t<32, 16, 16, 3, 3, 2, 1, 8, 8, 5><<<20480, 256, 0, stream>>>(actq, Apk, 4096);
  k_gemm_t<5, 2, 32, 64><<<512, 256, 0, stream>>>(Apk, Bpk, ab, actf, 4096);
  long long n3_4 = 1024LL * 32 * 8 * 8 / 4;
  k_absmax4<<<red_grid(n3_4), 256, 0, stream>>>((const float4*)actf, n3_4, slots + 4);
  k_quant4 <<<red_grid(n3_4), 256, 0, stream>>>((const float4*)actf, (unsigned*)actq, n3_4, slots + 4);

  k_prep<<<1, 256, 0, stream>>>(W[4], Bi[4], G[4], BB[4], MU[4], VAR[4],
                                1, 32, 288, 5, 2, slots + 4, Bpk, ab);
  k_pack_t<32, 8, 8, 3, 3, 1, 1, 8, 8, 5><<<20480, 256, 0, stream>>>(actq, Apk, 4096);
  k_gemm_t<5, 2, 32, 64><<<512, 256, 0, stream>>>(Apk, Bpk, ab, actf, 4096);
  long long n4_4 = 1024LL * 32 * 8 * 8 / 4;
  k_absmax4<<<red_grid(n4_4), 256, 0, stream>>>((const float4*)actf, n4_4, slots + 5);
  k_quant4 <<<red_grid(n4_4), 256, 0, stream>>>((const float4*)actf, (unsigned*)actq, n4_4, slots + 5);

  k_prep<<<1, 256, 0, stream>>>(wd, bd, nullptr, nullptr, nullptr, nullptr,
                                0, 10, 2048, 32, 1, slots + 5, Bpk, ab);
  k_pack_t<2048, 1, 1, 1, 1, 1, 0, 1, 1, 32><<<2048, 256, 0, stream>>>(actq, Apk, 64);
  k_gemm_t<32, 1, 10, 1><<<8, 256, 0, stream>>>(Apk, Bpk, ab, zbuf, 64);

  k_softmax<<<10, 256, 0, stream>>>(zbuf, (float*)d_out, 1024, 10);
}
